// FourOverSixGptOssMLP_73959336837605
// MI455X (gfx1250) — matrix operand order repaired and checked by reading, not compiled, not run on hardware
//
#include <hip/hip_runtime.h>
#include <stdint.h>

typedef __attribute__((ext_vector_type(16))) _Float16 v16h;
typedef __attribute__((ext_vector_type(8)))  float    v8f;
typedef __attribute__((ext_vector_type(2)))  _Float16 h2;

#define E_ 8
#define H_ 2880
#define I_ 2880
#define T_ 1024
#define K_ 4
#define QB 32
#define TM 64   // token rows per block tile
#define TN 64   // output cols per block tile
#define TK 32   // k-step (one wmma_f32_16x16x32_f16 depth)

__device__ __forceinline__ uint32_t h2_mul_bits(uint32_t a, uint32_t b) {
  union { uint32_t u; h2 h; } x, y, r;
  x.u = a; y.u = b;
  r.h = x.h * y.h;                 // v_pk_mul_f16
  return r.u;
}

// f16 bits of 2^(sa-12) in low half, 2^(sb-12) in high half (both normal f16).
__device__ __forceinline__ uint32_t pack_scale2(int sa, int sb) {
  return ((uint32_t)(sa + 3) << 10) | ((uint32_t)(sb + 3) << 26);
}

// Dequant 4 fp4 codes (each 0..15 in one int) with per-column e8m0 scales.
// lo = f16 pair for codes x,y ; hi = f16 pair for codes z,w. Exact.
__device__ __forceinline__ void fp4x4_dequant(int4 c, uint32_t sp01, uint32_t sp23,
                                              uint32_t& lo, uint32_t& hi) {
  uint32_t c4 = (uint32_t)c.x | ((uint32_t)c.y << 8) |
                ((uint32_t)c.z << 16) | ((uint32_t)c.w << 24);
  uint32_t m4 = c4 & 0x07070707u;      // magnitude codes
  uint32_t g4 = c4 & 0x08080808u;      // sign bits
  // byte table: f16 high bytes of {0,.5,1,1.5} / {2,3,4,6}
  uint32_t tb  = __builtin_amdgcn_perm(0x46444240u, 0x3E3C3800u, m4);
  uint32_t h01 = __builtin_amdgcn_perm(0u, tb, 0x01050004u);  // {0,tb0,0,tb1}
  uint32_t h23 = __builtin_amdgcn_perm(0u, tb, 0x03050204u);  // {0,tb2,0,tb3}
  uint32_t g01 = __builtin_amdgcn_perm(0u, g4, 0x01050004u);
  uint32_t g23 = __builtin_amdgcn_perm(0u, g4, 0x03050204u);
  lo = h2_mul_bits(h01 | (g01 << 4), sp01);   // sign 0x0800 -> 0x8000
  hi = h2_mul_bits(h23 | (g23 << 4), sp23);
}

__device__ __forceinline__ void store_h2(_Float16* p0, _Float16* p1, uint32_t v) {
  *(unsigned short*)p0 = (unsigned short)v;
  *(unsigned short*)p1 = (unsigned short)(v >> 16);
}

// Branchless round-to-nearest onto the fp4 grid (x already divided by scale).
__device__ __forceinline__ float fp4_round(float x) {
  float m = fabsf(x);
  int idx = (m >= 0.25f) + (m >= 0.75f) + (m >= 1.25f) + (m >= 1.75f) +
            (m >= 2.5f)  + (m >= 3.5f)  + (m >= 5.0f);
  int Ef = idx >> 1, M = idx & 1;
  int bits = (Ef == 0) ? (M * 0x3F000000) : (((126 + Ef) << 23) | (M << 22));
  return copysignf(__int_as_float(bits), x);
}

__global__ void zero_kernel(float* __restrict__ out, int n, int* __restrict__ cnt) {
  for (int i = blockIdx.x * blockDim.x + threadIdx.x; i < n; i += gridDim.x * blockDim.x)
    out[i] = 0.0f;
  if (blockIdx.x == 0 && threadIdx.x < E_) cnt[threadIdx.x] = 0;
}

__global__ void route_kernel(const int* __restrict__ ridx, const float* __restrict__ rw,
                             int* __restrict__ cnt, int* __restrict__ tlist,
                             float* __restrict__ clist) {
  int t = blockIdx.x * blockDim.x + threadIdx.x;
  if (t >= T_) return;
  for (int k = 0; k < K_; ++k) {
    int e = ridx[t * K_ + k];
    float w = rw[t * K_ + k];
    int slot = atomicAdd(&cnt[e], 1);
    if (slot < T_) {
      tlist[e * T_ + slot] = t;
      clist[e * T_ + slot] = w;
    }
  }
}

// MXFP4 fake-quant of activations, one thread per 32-block, exact f16 output.
__global__ void quantx_kernel(const float* __restrict__ x, _Float16* __restrict__ xq) {
  int b = blockIdx.x * blockDim.x + threadIdx.x;
  if (b >= (T_ * H_) / QB) return;
  const float4* p = (const float4*)(x + (size_t)b * QB);
  float v[QB];
  float amax = 0.0f;
#pragma unroll
  for (int i = 0; i < QB / 4; ++i) {
    float4 f = p[i];
    v[4*i+0] = f.x; v[4*i+1] = f.y; v[4*i+2] = f.z; v[4*i+3] = f.w;
  }
#pragma unroll
  for (int i = 0; i < QB; ++i) amax = fmaxf(amax, fabsf(v[i]));
  _Float16* o = xq + (size_t)b * QB;
  int eb = (__float_as_int(amax) >> 23) & 0xff;
  float sc  = __int_as_float((eb - 2) << 23);
  float inv = 1.0f / sc;
  bool nz = amax > 0.0f;
#pragma unroll
  for (int i = 0; i < QB; ++i)
    o[i] = (_Float16)(nz ? (fp4_round(v[i] * inv) * sc) : 0.0f);
}

// GEMM1: gate_up = xq @ W_gu (packed fp4 dequant), bias, clamp, gated SiLU,
// MXFP4 fake-quant along intermediate dim -> gact (f16).
// 2-stage software pipeline with double-buffered LDS (one barrier per k-step).
__global__ __launch_bounds__(256)
void gemm1_kernel(const _Float16* __restrict__ xq,
                  const int* __restrict__ codes, const int* __restrict__ scales,
                  const float* __restrict__ bias,
                  const int* __restrict__ cnt, const int* __restrict__ tlist,
                  _Float16* __restrict__ gact)
{
  const int e = blockIdx.z;
  const int n = cnt[e];
  const int s0 = blockIdx.x * TM;   // token tile fastest -> weight-tile L2 reuse
  if (s0 >= n) return;
  const int i0 = blockIdx.y * TN;

  __shared__ _Float16 Alds[2][TM][TK + 8];       // [buf][token][k]
  __shared__ _Float16 Blds[2][2 * TN][TK + 8];   // [buf][f local][k]
  __shared__ float    Glds[TM][TN + 8];          // gated activations
  __shared__ float    Slds[TM][2];               // per 32-block quant scales
  __shared__ int      toks[TM];

  const int tid  = threadIdx.x;
  const int lane = tid & 31;
  const int wave = tid >> 5;       // 8 waves
  const int rp   = wave >> 2;      // row pair: rows rp*16 and rp*16+32
  const int cq   = wave & 3;       // 16-col group

  if (tid < TM) {
    int s = s0 + tid;
    toks[tid] = tlist[e * T_ + (s < n ? s : 0)];
  }
  __syncthreads();

  v8f accg0 = {}, accu0 = {}, accg1 = {}, accu1 = {};

  // --- staging geometry ---
  const int myrow = tid >> 2;                   // A: 64 rows x 4 threads
  const int mycol = (tid & 3) << 3;             // 8 halfs per thread
  const _Float16* abase = xq + (size_t)toks[myrow] * H_ + mycol;

  const int ffq = (tid & 31) << 2;              // B: 4 consecutive f cols
  const int kkb = tid >> 5;                     // base k row, +8 per sub-iter
  const int64_t wbase = (int64_t)e * H_ * (2 * I_) + 2 * i0;
  const int64_t sbase = (int64_t)e * (H_ / QB) * (2 * I_) + 2 * i0;
  const int* cb = codes + wbase + ffq;
  const int* sb = scales + sbase + ffq;

  // --- fragment geometry ---
  const int lrow = lane & 15;
  const int ka = (lane >> 4) << 4;        // A frag: K base 0 or 16
  const int kb = (lane >> 4) << 4;        // B frag: K base 0 or 16
  const int arow0 = (rp << 4) + lrow;
  const int arow1 = arow0 + 32;
  const int fg = (((cq << 4) + lrow) << 1);

  // prologue: first tile into registers
  uint4 areg = *(const uint4*)abase;
  int4  sreg = *(const int4*)sb;
  int4  breg[4];
#pragma unroll
  for (int it = 0; it < 4; ++it)
    breg[it] = *(const int4*)(cb + (int64_t)(kkb + (it << 3)) * (2 * I_));
  cb += (int64_t)TK * (2 * I_);
  sb += 2 * I_;

  int p = 0;
  for (int k0 = 0; k0 < H_; k0 += TK) {
    // dequant registers -> LDS buffer p
    *(uint4*)&Alds[p][myrow][mycol] = areg;
    {
      uint32_t sp01 = pack_scale2(sreg.x, sreg.y);
      uint32_t sp23 = pack_scale2(sreg.z, sreg.w);
#pragma unroll
      for (int it = 0; it < 4; ++it) {
        int kk = kkb + (it << 3);
        uint32_t lo, hi;
        fp4x4_dequant(breg[it], sp01, sp23, lo, hi);
        store_h2(&Blds[p][ffq + 0][kk], &Blds[p][ffq + 1][kk], lo);
        store_h2(&Blds[p][ffq + 2][kk], &Blds[p][ffq + 3][kk], hi);
      }
    }
    // issue next tile's global loads (fly during the WMMA phase)
    if (k0 + TK < H_) {
      areg = *(const uint4*)(abase + k0 + TK);
      sreg = *(const int4*)sb;
#pragma unroll
      for (int it = 0; it < 4; ++it)
        breg[it] = *(const int4*)(cb + (int64_t)(kkb + (it << 3)) * (2 * I_));
      cb += (int64_t)TK * (2 * I_);
      sb += 2 * I_;
    }
    __syncthreads();

    v16h a0, a1, bg, bu;
#pragma unroll
    for (int j = 0; j < 8; ++j) {
      a0[j]     = Alds[p][arow0][ka + j];
      a0[j + 8] = Alds[p][arow0][ka + 8 + j];
      a1[j]     = Alds[p][arow1][ka + j];
      a1[j + 8] = Alds[p][arow1][ka + 8 + j];
    }
#pragma unroll
    for (int j = 0; j < 16; ++j) {
      bg[j] = Blds[p][fg][kb + j];
      bu[j] = Blds[p][fg + 1][kb + j];
    }
    accg0 = __builtin_amdgcn_wmma_f32_16x16x32_f16(false, a0, false, bg, (short)0, accg0, false, false);
    accu0 = __builtin_amdgcn_wmma_f32_16x16x32_f16(false, a0, false, bu, (short)0, accu0, false, false);
    accg1 = __builtin_amdgcn_wmma_f32_16x16x32_f16(false, a1, false, bg, (short)0, accg1, false, false);
    accu1 = __builtin_amdgcn_wmma_f32_16x16x32_f16(false, a1, false, bu, (short)0, accu1, false, false);
    p ^= 1;
  }
  __syncthreads();   // protect Glds (may alias activity) / quiesce before epilogue

  { // epilogue: bias + clamp + gated SiLU -> Glds (two row sets)
    int iglob = i0 + (cq << 4) + lrow;
    float bgv = bias[e * 2 * I_ + 2 * iglob];
    float buv = bias[e * 2 * I_ + 2 * iglob + 1];
    int nloc = (cq << 4) + lrow;
#pragma unroll
    for (int set = 0; set < 2; ++set) {
      v8f ag = set ? accg1 : accg0;
      v8f au = set ? accu1 : accu0;
      int rb = (rp << 4) + (set << 5);
#pragma unroll
      for (int r = 0; r < 8; ++r) {
        int mrow = rb + ((lane >> 4) << 3) + r;   // C layout: VGPR r -> M=r(+8)
        float g = ag[r] + bgv;
        float u = au[r] + buv;
        g = fminf(g, 7.0f);
        u = fminf(fmaxf(u, -7.0f), 7.0f);
        float act = (u + 1.0f) * (g / (1.0f + __expf(-1.702f * g)));
        Glds[mrow][nloc] = act;
      }
    }
  }
  __syncthreads();

  // MXFP4 fake-quant along i: 64 rows x 2 blocks of 32
  if (tid < 2 * TM) {
    int r = tid >> 1, b = tid & 1;
    float amax = 0.0f;
#pragma unroll 8
    for (int c = 0; c < QB; ++c) amax = fmaxf(amax, fabsf(Glds[r][(b << 5) + c]));
    float sc = 0.0f;
    if (amax > 0.0f) {
      int eb = (__float_as_int(amax) >> 23) & 0xff;
      sc = __int_as_float((eb - 2) << 23);
    }
    Slds[r][b] = sc;
  }
  __syncthreads();
#pragma unroll
  for (int it = 0; it < (TM * TN) / 256; ++it) {
    int idx = tid + (it << 8);
    int r = idx >> 6, c = idx & 63;
    int s = s0 + r;
    if (s < n) {
      float sc = Slds[r][c >> 5];
      float q = (sc > 0.0f) ? (fp4_round(Glds[r][c] / sc) * sc) : 0.0f;
      gact[((int64_t)e * T_ + s) * I_ + i0 + c] = (_Float16)q;
    }
  }
}

// GEMM2: out[t] += combine * (gact @ W_dn + bias), W_dn dequanted on the fly.
// Same 2-stage pipeline, double-buffered LDS.
__global__ __launch_bounds__(256)
void gemm2_kernel(const _Float16* __restrict__ gact,
                  const int* __restrict__ codes, const int* __restrict__ scales,
                  const float* __restrict__ bias,
                  const int* __restrict__ cnt, const int* __restrict__ tlist,
                  const float* __restrict__ clist,
                  float* __restrict__ out)
{
  const int e = blockIdx.z;
  const int n = cnt[e];
  const int s0 = blockIdx.x * TM;
  if (s0 >= n) return;
  const int h0 = blockIdx.y * TN;

  __shared__ _Float16 Alds[2][TM][TK + 8];
  __shared__ _Float16 Blds[2][TN][TK + 8];   // [buf][h local][k]
  __shared__ int      toks[TM];
  __shared__ float    comb[TM];

  const int tid  = threadIdx.x;
  const int lane = tid & 31;
  const int wave = tid >> 5;
  const int rp   = wave >> 2;
  const int cq   = wave & 3;

  if (tid < TM) {
    int s = s0 + tid;
    toks[tid] = tlist[e * T_ + (s < n ? s : 0)];
    comb[tid] = (s < n) ? clist[e * T_ + s] : 0.0f;
  }
  __syncthreads();

  v8f acc0 = {}, acc1 = {};

  const int myrow = tid >> 2;
  const int mycol = (tid & 3) << 3;
  const int sA = s0 + myrow;
  const _Float16* abase =
      gact + ((int64_t)e * T_ + (sA < n ? sA : s0)) * I_ + mycol;

  const int ffq = (tid & 15) << 2;       // 4 consecutive h cols
  const int kkr = tid >> 4;              // base k row, +16 per sub-iter
  const int64_t wbase = (int64_t)e * I_ * H_ + h0;
  const int64_t sbase = (int64_t)e * (I_ / QB) * H_ + h0;
  const int* cb = codes + wbase + ffq;
  const int* sb = scales + sbase + ffq;

  const int lrow = lane & 15;
  const int ka = (lane >> 4) << 4;
  const int kb = (lane >> 4) << 4;
  const int arow0 = (rp << 4) + lrow;
  const int arow1 = arow0 + 32;
  const int bcol = (cq << 4) + lrow;

  // prologue
  uint4 areg = *(const uint4*)abase;
  int4  sreg = *(const int4*)sb;
  int4  breg[2];
#pragma unroll
  for (int it = 0; it < 2; ++it)
    breg[it] = *(const int4*)(cb + (int64_t)(kkr + (it << 4)) * H_);
  cb += (int64_t)TK * H_;
  sb += H_;

  int p = 0;
  for (int k0 = 0; k0 < I_; k0 += TK) {
    *(uint4*)&Alds[p][myrow][mycol] = areg;
    {
      uint32_t sp01 = pack_scale2(sreg.x, sreg.y);
      uint32_t sp23 = pack_scale2(sreg.z, sreg.w);
#pragma unroll
      for (int it = 0; it < 2; ++it) {
        int kk = kkr + (it << 4);
        uint32_t lo, hi;
        fp4x4_dequant(breg[it], sp01, sp23, lo, hi);
        store_h2(&Blds[p][ffq + 0][kk], &Blds[p][ffq + 1][kk], lo);
        store_h2(&Blds[p][ffq + 2][kk], &Blds[p][ffq + 3][kk], hi);
      }
    }
    if (k0 + TK < I_) {
      areg = *(const uint4*)(abase + k0 + TK);
      sreg = *(const int4*)sb;
#pragma unroll
      for (int it = 0; it < 2; ++it)
        breg[it] = *(const int4*)(cb + (int64_t)(kkr + (it << 4)) * H_);
      cb += (int64_t)TK * H_;
      sb += H_;
    }
    __syncthreads();

    v16h a0, a1, b;
#pragma unroll
    for (int j = 0; j < 8; ++j) {
      a0[j]     = Alds[p][arow0][ka + j];
      a0[j + 8] = Alds[p][arow0][ka + 8 + j];
      a1[j]     = Alds[p][arow1][ka + j];
      a1[j + 8] = Alds[p][arow1][ka + 8 + j];
    }
#pragma unroll
    for (int j = 0; j < 16; ++j) b[j] = Blds[p][bcol][kb + j];
    acc0 = __builtin_amdgcn_wmma_f32_16x16x32_f16(false, a0, false, b, (short)0, acc0, false, false);
    acc1 = __builtin_amdgcn_wmma_f32_16x16x32_f16(false, a1, false, b, (short)0, acc1, false, false);
    p ^= 1;
  }

  int h = h0 + (cq << 4) + lrow;
  float bv = bias[e * H_ + h];
#pragma unroll
  for (int set = 0; set < 2; ++set) {
    v8f acc = set ? acc1 : acc0;
    int rb = (rp << 4) + (set << 5);
#pragma unroll
    for (int r = 0; r < 8; ++r) {
      int mloc = rb + ((lane >> 4) << 3) + r;
      int s = s0 + mloc;
      if (s < n) {
        float val = (acc[r] + bv) * comb[mloc];
        atomicAdd(&out[(size_t)toks[mloc] * H_ + h], val);
      }
    }
  }
}

extern "C" void kernel_launch(void* const* d_in, const int* in_sizes, int n_in,
                              void* d_out, int out_size, void* d_ws, size_t ws_size,
                              hipStream_t stream)
{
  (void)in_sizes; (void)n_in; (void)out_size; (void)ws_size;
  const float* x    = (const float*)d_in[0];
  const int*   ridx = (const int*)d_in[1];
  const float* rw   = (const float*)d_in[2];
  const int*   guc  = (const int*)d_in[3];
  const int*   gus  = (const int*)d_in[4];
  const float* gub  = (const float*)d_in[5];
  const int*   dnc  = (const int*)d_in[6];
  const int*   dns  = (const int*)d_in[7];
  const float* dnb  = (const float*)d_in[8];
  float* out = (float*)d_out;

  char* ws = (char*)d_ws;
  size_t off = 0;
  auto alloc = [&](size_t bytes) -> char* {
    off = (off + 255) & ~(size_t)255;
    char* p = ws + off;
    off += bytes;
    return p;
  };
  int*      cnt   = (int*)alloc(E_ * sizeof(int));
  int*      tlist = (int*)alloc((size_t)E_ * T_ * sizeof(int));
  float*    clist = (float*)alloc((size_t)E_ * T_ * sizeof(float));
  _Float16* xq    = (_Float16*)alloc((size_t)T_ * H_ * sizeof(_Float16));
  _Float16* gact  = (_Float16*)alloc((size_t)E_ * T_ * I_ * sizeof(_Float16));

  zero_kernel<<<512, 256, 0, stream>>>(out, T_ * H_, cnt);
  route_kernel<<<(T_ + 255) / 256, 256, 0, stream>>>(ridx, rw, cnt, tlist, clist);
  quantx_kernel<<<((T_ * H_ / QB) + 255) / 256, 256, 0, stream>>>(x, xq);

  dim3 g1(T_ / TM, I_ / TN, E_);   // token tile fastest for weight L2 reuse
  gemm1_kernel<<<g1, 256, 0, stream>>>(xq, guc, gus, gub, cnt, tlist, gact);

  dim3 g2(T_ / TM, H_ / TN, E_);
  gemm2_kernel<<<g2, 256, 0, stream>>>(gact, dnc, dns, dnb, cnt, tlist, clist, out);
}
